// TemporalGNNEncoder_28853590295053
// MI455X (gfx1250) — matrix operand order repaired and checked by reading, not compiled, not run on hardware
//
#include <hip/hip_runtime.h>

// ---------------------------------------------------------------------------
// TemporalGNNEncoder for MI455X (gfx1250, wave32, WMMA bf16 16x16x32)
// ---------------------------------------------------------------------------
#define NNODES 127488
#define NEDGES 2039808
#define HDIM   128
#define INDIM  5
#define BGRAPH 512
#define CCH    83
#define TSTEP  3
#define MROWS  42496          // B*C sequences for the GRU
#define EMBD   64
#define BN_EPS 1e-5f

typedef __attribute__((ext_vector_type(16))) __bf16 v16bf;
typedef __attribute__((ext_vector_type(8)))  __bf16 v8bf;
typedef __attribute__((ext_vector_type(8)))  float  v8f;

union AFrag { v16bf v; v8bf h[2]; __bf16 e[16]; };

__device__ __forceinline__ v8f wmma_bf16(v16bf a, v16bf b, v8f c) {
  // D = A(16x32 bf16) x B(32x16 bf16) + C(16x16 f32)
  return __builtin_amdgcn_wmma_f32_16x16x32_bf16(false, a, false, b,
                                                 (short)0, c, false, false);
}

// ---------------------------------------------------------------------------
// small utility kernels
// ---------------------------------------------------------------------------
__global__ void k_fill(float* p, int n, float v) {
  int i = blockIdx.x * 256 + threadIdx.x;
  if (i < n) p[i] = v;
}

__global__ void k_deg_count(const int* col, float* deg, int e) {
  int i = blockIdx.x * 256 + threadIdx.x;
  if (i < e) atomicAdd(&deg[col[i]], 1.0f);
}

__global__ void k_rsqrt(float* d, int n) {
  int i = blockIdx.x * 256 + threadIdx.x;
  if (i < n) d[i] = rsqrtf(d[i]);
}

// Repack weight matrix into WMMA B-fragment order (dense 32x16 bf16 layout:
// lane L holds column n=L&15, K = (L>>4)*8 + i for i<8, 16 + (L>>4)*8 + (i-8) for i>=8, per 32-k block).
// transposed==0 : B(k,n) = W[k*ld + n]   (conv weights, emb_w: (K,N) row-major)
// transposed==1 : B(k,n) = W[n*ld + k]   (GRU w_ih/w_hh: (N,K) row-major)
__global__ void k_repack(const float* W, __bf16* frag, int K, int Nw,
                         int transposed, int ld) {
  int tid = blockIdx.x * 256 + threadIdx.x;
  int KB = K / 32;
  int total = (Nw / 16) * KB * 32;
  if (tid >= total) return;
  int lane = tid & 31;
  int kb = (tid >> 5) % KB;
  int nt = tid / (32 * KB);
  int n = nt * 16 + (lane & 15);
  int kbase = kb * 32 + (lane >> 4) * 8;
#pragma unroll
  for (int i = 0; i < 16; i++) {
    int k = kbase + (i < 8 ? i : i + 8);
    float v = transposed ? W[n * ld + k] : W[k * ld + n];
    frag[tid * 16 + i] = (__bf16)v;
  }
}

// ---------------------------------------------------------------------------
// GCN layer pieces
// ---------------------------------------------------------------------------
// layer-0 linear: (N,5)@(5,128) -> tiny, scalar path
__global__ void k_lin0(const float* x, const float* W0, float* lin) {
  int idx = blockIdx.x * 256 + threadIdx.x;
  if (idx >= NNODES * HDIM) return;
  int n = idx >> 7, j = idx & 127;
  float s = 0.f;
#pragma unroll
  for (int k = 0; k < INDIM; k++) s += x[n * INDIM + k] * W0[k * HDIM + j];
  lin[idx] = s;
}

// GEMM (N x 128) @ (128 x 128) via WMMA; 8 waves/block = 8 N-tiles, 16 rows/block
__global__ void k_gemm_gcn(const __bf16* A, const __bf16* frag, float* out) {
  int lane = threadIdx.x & 31;
  int nt = threadIdx.x >> 5;                     // 0..7 -> N tile
  int m = (blockIdx.x << 4) + (lane & 15);
  int koff = (lane >> 4) << 3;
  const __bf16* arow = A + (size_t)m * HDIM;
  v8f acc = {};
#pragma unroll
  for (int kb = 0; kb < 4; kb++) {
    AFrag a;
    a.h[0] = *(const v8bf*)(arow + kb * 32 + koff);
    a.h[1] = *(const v8bf*)(arow + kb * 32 + 16 + koff);
    v16bf b = *(const v16bf*)(frag + ((size_t)(nt * 4 + kb) * 32 + lane) * 16);
    acc = wmma_bf16(a.v, b, acc);
  }
  int col = (nt << 4) + (lane & 15);
  int row0 = (blockIdx.x << 4) + ((lane >> 4) << 3);
#pragma unroll
  for (int r = 0; r < 8; r++)
    out[(size_t)(row0 + r) * HDIM + col] = acc[r];
}

// agg = lin * dinv^2 + bias  (self-loop term, initializes the scatter target)
__global__ void k_selfinit(const float* lin, const float* dinv,
                           const float* bias, float* agg) {
  int idx = blockIdx.x * 256 + threadIdx.x;
  if (idx >= NNODES * HDIM) return;
  int n = idx >> 7, j = idx & 127;
  float dn = dinv[n];
  agg[idx] = lin[idx] * dn * dn + bias[j];
}

// one wave32 per edge, 4 floats/lane; L2-resident atomics
__global__ void k_scatter(const float* lin, float* agg, const int* row,
                          const int* col, const float* dinv) {
  int e = blockIdx.x * 8 + (threadIdx.x >> 5);
  int lane = threadIdx.x & 31;
  if (e >= NEDGES) return;
  int r = row[e], c = col[e];
  float nrm = dinv[r] * dinv[c];
  const float4 v = *(const float4*)(lin + (size_t)r * HDIM + lane * 4);
  float* dst = agg + (size_t)c * HDIM + lane * 4;
  atomicAdd(dst + 0, v.x * nrm);
  atomicAdd(dst + 1, v.y * nrm);
  atomicAdd(dst + 2, v.z * nrm);
  atomicAdd(dst + 3, v.w * nrm);
}

// per-column sum / sumsq -> stats[0:128], stats[128:256]
__global__ void k_bnstats(const float* h, float* stats) {
  __shared__ float ls[256], lq[256];
  int tid = threadIdx.x;
  int j = tid & 127, rg = tid >> 7;
  int chunk = (NNODES + gridDim.x - 1) / gridDim.x;
  int start = blockIdx.x * chunk;
  int end = start + chunk;
  if (end > NNODES) end = NNODES;
  float s = 0.f, q = 0.f;
  for (int r = start + rg; r < end; r += 2) {
    float v = h[(size_t)r * HDIM + j];
    s += v; q += v * v;
  }
  ls[tid] = s; lq[tid] = q;
  __syncthreads();
  if (tid < 128) {
    atomicAdd(&stats[j], ls[tid] + ls[tid + 128]);
    atomicAdd(&stats[128 + j], lq[tid] + lq[tid + 128]);
  }
}

// BN (batch stats, biased var) + ReLU, emit bf16 activations for next GEMM
__global__ void k_bnapply(const float* h, const float* stats,
                          const float* gamma, const float* beta, __bf16* hbf) {
  int idx = blockIdx.x * 256 + threadIdx.x;
  if (idx >= NNODES * HDIM) return;
  int j = idx & 127;
  const float invN = 1.0f / (float)NNODES;
  float mean = stats[j] * invN;
  float var = stats[128 + j] * invN - mean * mean;
  float sc = gamma[j] * rsqrtf(var + BN_EPS);
  float v = (h[idx] - mean) * sc + beta[j];
  hbf[idx] = (__bf16)fmaxf(v, 0.f);
}

// ---------------------------------------------------------------------------
// GRU
// ---------------------------------------------------------------------------
// gi = x_t @ w_ih^T + b_ih ; rows gathered by node index (b,c,t) mapping.
// 8 waves * 3 N-tiles = 24 tiles = 384 cols.
__global__ void k_gi(const __bf16* hbf, const __bf16* frag, const float* b_ih,
                     float* gi, int t) {
  int lane = threadIdx.x & 31;
  int wave = threadIdx.x >> 5;
  int mrow = (blockIdx.x << 4) + (lane & 15);
  int b = mrow / CCH;
  int c = mrow - b * CCH;
  int node = (b * TSTEP + t) * CCH + c;
  int koff = (lane >> 4) << 3;
  const __bf16* arow = hbf + (size_t)node * HDIM;
  AFrag a[4];
#pragma unroll
  for (int kb = 0; kb < 4; kb++) {
    a[kb].h[0] = *(const v8bf*)(arow + kb * 32 + koff);
    a[kb].h[1] = *(const v8bf*)(arow + kb * 32 + 16 + koff);
  }
  int row0 = (blockIdx.x << 4) + ((lane >> 4) << 3);
#pragma unroll
  for (int s = 0; s < 3; s++) {
    int nt = wave * 3 + s;
    v8f acc = {};
#pragma unroll
    for (int kb = 0; kb < 4; kb++) {
      v16bf bb = *(const v16bf*)(frag + ((size_t)(nt * 4 + kb) * 32 + lane) * 16);
      acc = wmma_bf16(a[kb].v, bb, acc);
    }
    int col = (nt << 4) + (lane & 15);
    float bias = b_ih[col];
#pragma unroll
    for (int r = 0; r < 8; r++)
      gi[(size_t)(row0 + r) * 384 + col] = acc[r] + bias;
  }
}

// gh = hprev @ w_hh^T (WMMA, via LDS) fused with gate math; hprev updated in place
__global__ void k_gru_step(float* hprev, const float* gi, const float* b_hh,
                           const __bf16* fragHH) {
  __shared__ float ghs[16 * 384];                // 24 KB
  int lane = threadIdx.x & 31;
  int wave = threadIdx.x >> 5;
  int row = (blockIdx.x << 4) + (lane & 15);
  int koff = (lane >> 4) << 3;
  const float* hrow = hprev + (size_t)row * HDIM;
  AFrag a[4];
#pragma unroll
  for (int kb = 0; kb < 4; kb++) {
#pragma unroll
    for (int i = 0; i < 8; i++) a[kb].e[i] = (__bf16)hrow[kb * 32 + koff + i];
#pragma unroll
    for (int i = 0; i < 8; i++) a[kb].e[8 + i] = (__bf16)hrow[kb * 32 + 16 + koff + i];
  }
  int rows0 = (lane >> 4) << 3;
#pragma unroll
  for (int s = 0; s < 3; s++) {
    int nt = wave * 3 + s;
    v8f acc = {};
#pragma unroll
    for (int kb = 0; kb < 4; kb++) {
      v16bf bb = *(const v16bf*)(fragHH + ((size_t)(nt * 4 + kb) * 32 + lane) * 16);
      acc = wmma_bf16(a[kb].v, bb, acc);
    }
    int col = (nt << 4) + (lane & 15);
#pragma unroll
    for (int r = 0; r < 8; r++) ghs[(rows0 + r) * 384 + col] = acc[r];
  }
  __syncthreads();
#pragma unroll
  for (int it = 0; it < 8; it++) {
    int idx = threadIdx.x + it * 256;            // 16 rows x 128 cols
    int mm = idx >> 7, j = idx & 127;
    int grow = (blockIdx.x << 4) + mm;
    float gir = gi[(size_t)grow * 384 + j];
    float giz = gi[(size_t)grow * 384 + 128 + j];
    float gin = gi[(size_t)grow * 384 + 256 + j];
    float ghr = ghs[mm * 384 + j] + b_hh[j];
    float ghz = ghs[mm * 384 + 128 + j] + b_hh[128 + j];
    float ghn = ghs[mm * 384 + 256 + j] + b_hh[256 + j];
    float rr = 1.f / (1.f + __expf(-(gir + ghr)));
    float zz = 1.f / (1.f + __expf(-(giz + ghz)));
    float nn = tanhf(gin + rr * ghn);
    float hp = hprev[(size_t)grow * HDIM + j];
    hprev[(size_t)grow * HDIM + j] = (1.f - zz) * nn + zz * hp;
  }
}

// mean over C channels, emit bf16 for final projection
__global__ void k_mean(const float* hT, __bf16* gbf) {
  int idx = blockIdx.x * 256 + threadIdx.x;
  if (idx >= BGRAPH * HDIM) return;
  int b = idx >> 7, j = idx & 127;
  float s = 0.f;
  for (int c = 0; c < CCH; c++) s += hT[(size_t)(b * CCH + c) * HDIM + j];
  gbf[idx] = (__bf16)(s * (1.0f / (float)CCH));
}

// final projection (512x128)@(128x64)+b -> d_out ; 4 waves = 4 N-tiles
__global__ void k_emb(const __bf16* gbf, const __bf16* frag, const float* eb,
                      float* out) {
  int lane = threadIdx.x & 31;
  int nt = threadIdx.x >> 5;                     // 0..3
  int m = (blockIdx.x << 4) + (lane & 15);
  int koff = (lane >> 4) << 3;
  const __bf16* arow = gbf + (size_t)m * HDIM;
  v8f acc = {};
#pragma unroll
  for (int kb = 0; kb < 4; kb++) {
    AFrag a;
    a.h[0] = *(const v8bf*)(arow + kb * 32 + koff);
    a.h[1] = *(const v8bf*)(arow + kb * 32 + 16 + koff);
    v16bf b = *(const v16bf*)(frag + ((size_t)(nt * 4 + kb) * 32 + lane) * 16);
    acc = wmma_bf16(a.v, b, acc);
  }
  int col = (nt << 4) + (lane & 15);
  float bias = eb[col];
  int row0 = (blockIdx.x << 4) + ((lane >> 4) << 3);
#pragma unroll
  for (int r = 0; r < 8; r++)
    out[(size_t)(row0 + r) * EMBD + col] = acc[r] + bias;
}

// ---------------------------------------------------------------------------
// workspace layout (all offsets 256B aligned)
// ---------------------------------------------------------------------------
constexpr size_t OFF_DINV  = 0;                                   // N f32
constexpr size_t OFF_LIN   = OFF_DINV + (size_t)NNODES * 4;       // N*128 f32 (also gi)
constexpr size_t OFF_AGG   = OFF_LIN + (size_t)NNODES * HDIM * 4; // N*128 f32 (also hprev)
constexpr size_t OFF_HBF   = OFF_AGG + (size_t)NNODES * HDIM * 4; // N*128 bf16
constexpr size_t OFF_STATS = OFF_HBF + (size_t)NNODES * HDIM * 2; // 256 f32
constexpr size_t OFF_FW1   = OFF_STATS + 1024;                    // 32 KB
constexpr size_t OFF_FW2   = OFF_FW1 + 32768;                     // 32 KB
constexpr size_t OFF_FIH   = OFF_FW2 + 32768;                     // 96 KB
constexpr size_t OFF_FHH   = OFF_FIH + 98304;                     // 96 KB
constexpr size_t OFF_FEMB  = OFF_FHH + 98304;                     // 16 KB
constexpr size_t OFF_GBF   = OFF_FEMB + 16384;                    // 512*128 bf16

extern "C" void kernel_launch(void* const* d_in, const int* in_sizes, int n_in,
                              void* d_out, int out_size, void* d_ws, size_t ws_size,
                              hipStream_t stream) {
  (void)in_sizes; (void)n_in; (void)out_size; (void)ws_size;
  // setup_inputs() leaves in dict order
  const float* x     = (const float*)d_in[0];
  const int*   ei    = (const int*)d_in[1];          // [2,E]
  /* batch d_in[2] unused (graphs are contiguous) */
  const float* W0    = (const float*)d_in[3];
  const float* W1    = (const float*)d_in[4];
  const float* W2    = (const float*)d_in[5];
  const float* cb[3] = {(const float*)d_in[6], (const float*)d_in[7], (const float*)d_in[8]};
  const float* bg[3] = {(const float*)d_in[9], (const float*)d_in[10], (const float*)d_in[11]};
  const float* bb[3] = {(const float*)d_in[12], (const float*)d_in[13], (const float*)d_in[14]};
  const float* w_ih  = (const float*)d_in[15];
  const float* w_hh  = (const float*)d_in[16];
  const float* b_ih  = (const float*)d_in[17];
  const float* b_hh  = (const float*)d_in[18];
  const float* emb_w = (const float*)d_in[19];
  const float* emb_b = (const float*)d_in[20];
  float* out = (float*)d_out;

  char* ws = (char*)d_ws;
  float*  dinv    = (float*)(ws + OFF_DINV);
  float*  bufLin  = (float*)(ws + OFF_LIN);
  float*  bufAgg  = (float*)(ws + OFF_AGG);
  __bf16* hbf     = (__bf16*)(ws + OFF_HBF);
  float*  stats   = (float*)(ws + OFF_STATS);
  __bf16* fragW1  = (__bf16*)(ws + OFF_FW1);
  __bf16* fragW2  = (__bf16*)(ws + OFF_FW2);
  __bf16* fragIH  = (__bf16*)(ws + OFF_FIH);
  __bf16* fragHH  = (__bf16*)(ws + OFF_FHH);
  __bf16* fragEMB = (__bf16*)(ws + OFF_FEMB);
  __bf16* gbf     = (__bf16*)(ws + OFF_GBF);

  const int* erow = ei;
  const int* ecol = ei + NEDGES;
  const int NH_BLOCKS = (NNODES * HDIM + 255) / 256;

  // degree / normalization (self-loop baked in as 1.0 init)
  k_fill<<<(NNODES + 255) / 256, 256, 0, stream>>>(dinv, NNODES, 1.0f);
  k_deg_count<<<(NEDGES + 255) / 256, 256, 0, stream>>>(ecol, dinv, NEDGES);
  k_rsqrt<<<(NNODES + 255) / 256, 256, 0, stream>>>(dinv, NNODES);

  // one-time weight repacks into WMMA B-fragment order
  k_repack<<<4, 256, 0, stream>>>(W1, fragW1, 128, 128, 0, 128);
  k_repack<<<4, 256, 0, stream>>>(W2, fragW2, 128, 128, 0, 128);
  k_repack<<<12, 256, 0, stream>>>(w_ih, fragIH, 128, 384, 1, 128);
  k_repack<<<12, 256, 0, stream>>>(w_hh, fragHH, 128, 384, 1, 128);
  k_repack<<<2, 256, 0, stream>>>(emb_w, fragEMB, 128, 64, 0, 64);

  // 3 GCN layers
  for (int l = 0; l < 3; l++) {
    if (l == 0)
      k_lin0<<<NH_BLOCKS, 256, 0, stream>>>(x, W0, bufLin);
    else
      k_gemm_gcn<<<NNODES / 16, 256, 0, stream>>>(hbf, (l == 1 ? fragW1 : fragW2), bufLin);
    k_selfinit<<<NH_BLOCKS, 256, 0, stream>>>(bufLin, dinv, cb[l], bufAgg);
    k_scatter<<<NEDGES / 8, 256, 0, stream>>>(bufLin, bufAgg, erow, ecol, dinv);
    k_fill<<<1, 256, 0, stream>>>(stats, 256, 0.0f);
    k_bnstats<<<128, 256, 0, stream>>>(bufAgg, stats);
    k_bnapply<<<NH_BLOCKS, 256, 0, stream>>>(bufAgg, stats, bg[l], bb[l], hbf);
  }

  // GRU over T=3 (hprev aliases bufAgg, gi aliases bufLin — both dead now)
  float* hprev = bufAgg;
  float* gi = bufLin;
  k_fill<<<(MROWS * HDIM + 255) / 256, 256, 0, stream>>>(hprev, MROWS * HDIM, 0.0f);
  for (int t = 0; t < TSTEP; t++) {
    k_gi<<<MROWS / 16, 256, 0, stream>>>(hbf, fragIH, b_ih, gi, t);
    k_gru_step<<<MROWS / 16, 256, 0, stream>>>(hprev, gi, b_hh, fragHH);
  }

  // channel-mean pool + final projection
  k_mean<<<(BGRAPH * HDIM + 255) / 256, 256, 0, stream>>>(hprev, gbf);
  k_emb<<<BGRAPH / 16, 128, 0, stream>>>(gbf, fragEMB, emb_b, out);
}
